// FluxAttention_67164698574891
// MI455X (gfx1250) — matrix operand order repaired and checked by reading, not compiled, not run on hardware
//
#include <hip/hip_runtime.h>
#include <hip/hip_bf16.h>

typedef __bf16 bf16;
typedef __attribute__((ext_vector_type(8)))  __bf16 v8bf;
typedef __attribute__((ext_vector_type(16))) __bf16 v16bf;
typedef __attribute__((ext_vector_type(8)))  float  v8f;
typedef __attribute__((ext_vector_type(4)))  unsigned int v4u;
typedef __attribute__((ext_vector_type(8)))  int v8i;
typedef __attribute__((ext_vector_type(4)))  int v4i;

constexpr int TXT = 256, IMG = 2304, SEQ = 2560;
constexpr int DD = 3072, NH = 24, DHD = 128, E3 = 9216;
constexpr float EPS = 1e-5f;
constexpr float SCALE = 0.08838834764831845f; // 1/sqrt(128)

#if defined(__has_builtin)
#if __has_builtin(__builtin_amdgcn_tensor_load_to_lds) && \
    __has_builtin(__builtin_amdgcn_s_wait_tensorcnt)
#define USE_TDM 1
#endif
#endif
#ifndef USE_TDM
#define USE_TDM 0
#endif

static __device__ __forceinline__ v16bf cat8(v8bf lo, v8bf hi) {
  v16bf r;
#pragma unroll
  for (int i = 0; i < 8; ++i) { r[i] = lo[i]; r[i + 8] = hi[i]; }
  return r;
}
static __device__ __forceinline__ v8f vzero8() {
  v8f r;
#pragma unroll
  for (int i = 0; i < 8; ++i) r[i] = 0.0f;
  return r;
}

#if USE_TDM
// Issue a TDM 2D tile load: tile_d0 (contiguous, elems) x tile_d1 rows of bf16,
// row stride = stride0 elems, into LDS at byte offset lds_off.
// D# layout per CDNA5 ISA 8.3-8.5 (group0 128b, group1 256b; groups 2/3 zero).
// This toolchain's builtin arity: (v4u g0, v8i g1, v4i g2, v4i g3, v8i, i32 cpol).
static __device__ __forceinline__ void tdm_load_2d(const bf16* g, unsigned lds_off,
                                                   int tile_d0, int tile_d1,
                                                   long long stride0) {
  const unsigned long long ga = (unsigned long long)(size_t)g;
  v4u g0;
  g0[0] = 1u;                                   // count=1 (valid), user mode
  g0[1] = lds_off;                              // lds_addr (bytes)
  g0[2] = (unsigned)(ga & 0xffffffffu);         // global_addr[31:0]
  g0[3] = (unsigned)((ga >> 32) & 0x1ffffffu)   // global_addr[56:32]
        | (2u << 30);                           // type=2 ("image")
  v8i g1;
  g1[0] = (int)(1u << 16);                      // data_size=1 -> 2 bytes
  g1[1] = (int)(((unsigned)tile_d0 & 0xffffu) << 16);          // tensor_dim0[15:0]
  g1[2] = (int)((((unsigned)tile_d0 >> 16) & 0xffffu)
        | (((unsigned)tile_d1 & 0xffffu) << 16));              // td0[31:16]|td1[15:0]
  g1[3] = (int)((((unsigned)tile_d1 >> 16) & 0xffffu)
        | (((unsigned)tile_d0 & 0xffffu) << 16));              // td1[31:16]|tile_dim0
  g1[4] = (int)((unsigned)tile_d1 & 0xffffu);                  // tile_dim1 (tile_dim2=0)
  g1[5] = (int)(unsigned)(stride0 & 0xffffffffll);             // dim0_stride[31:0]
  g1[6] = (int)(unsigned)((stride0 >> 32) & 0xffffll);         // dim0_stride[47:32]
  g1[7] = 0;
  const v4i z4 = {0, 0, 0, 0};
  const v8i z8 = {0, 0, 0, 0, 0, 0, 0, 0};
  __builtin_amdgcn_tensor_load_to_lds(g0, g1, z4, z4, z8, 0);
}
#endif

// ---------------------------------------------------------------- f32 -> bf16
__global__ __launch_bounds__(256) void cvt_bf16(const float* __restrict__ x,
                                                bf16* __restrict__ y, size_t n) {
  size_t i = (size_t)blockIdx.x * blockDim.x + threadIdx.x;
  size_t stride = (size_t)gridDim.x * blockDim.x;
  for (; i < n; i += stride) y[i] = (bf16)x[i];
}

// ---------------------------------------------------------------- WMMA GEMM
// C[M,N] = A[M,K] * W[N,K]^T (+ bias[N]).  A,W bf16 row-major (K contiguous).
// Block = 256 thr (8 waves, 4x2), tile 128x128, K-step 64.  A/B tiles staged
// in LDS by the Tensor Data Mover (double buffered), fragments ds_load'ed.
__global__ __launch_bounds__(256) void gemm_bf16(
    const bf16* __restrict__ A, const bf16* __restrict__ W,
    const float* __restrict__ bias, float* __restrict__ C,
    int M, int N, int K) {
  __shared__ bf16 tA[2][128 * 64];
  __shared__ bf16 tB[2][128 * 64];
  const int tid = threadIdx.x;
  const int lane = tid & 31, wave = tid >> 5;
  const int l16 = lane & 15, hi = lane >> 4;
  const int mw = (wave >> 1) * 32, nw = (wave & 1) * 64;   // 4x2 wave grid
  const int mblk = blockIdx.y * 128, nblk = blockIdx.x * 128;

  v8f acc[2][4];
#pragma unroll
  for (int i = 0; i < 2; ++i)
#pragma unroll
    for (int j = 0; j < 4; ++j) acc[i][j] = vzero8();

  auto compute = [&](const bf16* pA, const bf16* pB) {
#pragma unroll
    for (int kk = 0; kk < 64; kk += 32) {
      v16bf a[2], b[4];
#pragma unroll
      for (int i = 0; i < 2; ++i) {   // A frag: lane=row, 16 contiguous K @ hi*16
        const bf16* ap = pA + (mw + i * 16 + l16) * 64 + kk;
        a[i] = cat8(*(const v8bf*)(ap + hi * 16),
                    *(const v8bf*)(ap + hi * 16 + 8));
      }
#pragma unroll
      for (int j = 0; j < 4; ++j) {   // B frag: lane=col, 16 contiguous K @ hi*16
        const bf16* bp = pB + (nw + j * 16 + l16) * 64 + kk + hi * 16;
        b[j] = cat8(*(const v8bf*)bp, *(const v8bf*)(bp + 8));
      }
#pragma unroll
      for (int i = 0; i < 2; ++i)
#pragma unroll
        for (int j = 0; j < 4; ++j)
          acc[i][j] = __builtin_amdgcn_wmma_f32_16x16x32_bf16(
              false, a[i], false, b[j], (short)0, acc[i][j], false, false);
    }
  };

  const int ksteps = K / 64;
#if USE_TDM
  if (wave == 0) {  // wave 0 drives the TDM (EXEC-independent, wave-level DMA)
    tdm_load_2d(A + (size_t)mblk * K, (unsigned)(size_t)&tA[0][0], 64, 128, K);
    tdm_load_2d(W + (size_t)nblk * K, (unsigned)(size_t)&tB[0][0], 64, 128, K);
  }
  for (int ks = 0; ks < ksteps; ++ks) {
    const int buf = ks & 1;
    if (wave == 0) {
      if (ks + 1 < ksteps) {  // prefetch next tile into the other buffer
        const int k1 = (ks + 1) * 64;
        tdm_load_2d(A + (size_t)mblk * K + k1,
                    (unsigned)(size_t)&tA[buf ^ 1][0], 64, 128, K);
        tdm_load_2d(W + (size_t)nblk * K + k1,
                    (unsigned)(size_t)&tB[buf ^ 1][0], 64, 128, K);
        __builtin_amdgcn_s_wait_tensorcnt((short)2);  // current tile done
      } else {
        __builtin_amdgcn_s_wait_tensorcnt((short)0);
      }
    }
    __syncthreads();          // publish tile to all waves
    compute(tA[buf], tB[buf]);
    __syncthreads();          // tile consumed; safe to overwrite
  }
#else
  for (int ks = 0; ks < ksteps; ++ks) {
    const int k0 = ks * 64;
    for (int c = tid; c < 128 * 64 / 8; c += 256) {  // cooperative fallback copy
      const int r = c >> 3, kk = (c & 7) * 8;
      *(v8bf*)&tA[0][r * 64 + kk] =
          *(const v8bf*)(A + (size_t)(mblk + r) * K + k0 + kk);
      *(v8bf*)&tB[0][r * 64 + kk] =
          *(const v8bf*)(W + (size_t)(nblk + r) * K + k0 + kk);
    }
    __syncthreads();
    compute(tA[0], tB[0]);
    __syncthreads();
  }
#endif

#pragma unroll
  for (int i = 0; i < 2; ++i)
#pragma unroll
    for (int j = 0; j < 4; ++j) {
      const int col = nblk + nw + j * 16 + l16;
      const float bj = bias ? bias[col] : 0.0f;
#pragma unroll
      for (int r = 0; r < 8; ++r) {
        const int row = mblk + mw + i * 16 + hi * 8 + r;
        C[(size_t)row * N + col] = acc[i][j][r] + bj;
      }
    }
}

// ------------------------------------------------- RMSNorm + RoPE + pack QKV
// One wave per (s, h).  Emits Q,K as [H,S,DH] bf16, V transposed [H,DH,S].
__global__ __launch_bounds__(256) void pack_qkv(
    const float* __restrict__ qkv_img, const float* __restrict__ qkv_txt,
    const float* __restrict__ cosb, const float* __restrict__ sinb,
    const float* __restrict__ nq, const float* __restrict__ nk,
    const float* __restrict__ naq, const float* __restrict__ nak,
    bf16* __restrict__ Qm, bf16* __restrict__ Km, bf16* __restrict__ Vt) {
  const int lane = threadIdx.x & 31, wv = threadIdx.x >> 5;
  const int task = blockIdx.x * 8 + wv;
  const int s = task / NH, h = task % NH;

  const float* src; const float* qw; const float* kw; int srow;
  if (s < TXT) { src = qkv_txt; srow = s;       qw = naq; kw = nak; }
  else         { src = qkv_img; srow = s - TXT; qw = nq;  kw = nk;  }
  const float* base = src + (size_t)srow * E3 + h * DHD;
  const int d0 = lane * 4;

  float q[4], k[4], v[4];
#pragma unroll
  for (int t = 0; t < 4; ++t) {
    q[t] = base[d0 + t];
    k[t] = base[DD + d0 + t];
    v[t] = base[2 * DD + d0 + t];
  }
  float sq = 0.f, sk = 0.f;
#pragma unroll
  for (int t = 0; t < 4; ++t) { sq += q[t] * q[t]; sk += k[t] * k[t]; }
#pragma unroll
  for (int msk = 1; msk < 32; msk <<= 1) {
    sq += __shfl_xor(sq, msk, 32);
    sk += __shfl_xor(sk, msk, 32);
  }
  const float rq = rsqrtf(sq * (1.0f / DHD) + EPS);
  const float rk = rsqrtf(sk * (1.0f / DHD) + EPS);
#pragma unroll
  for (int t = 0; t < 4; ++t) { q[t] *= rq * qw[d0 + t]; k[t] *= rk * kw[d0 + t]; }

#pragma unroll
  for (int p = 0; p < 2; ++p) {  // RoPE on pairs (2i, 2i+1); i = lane*2+p
    const int i = lane * 2 + p;
    const float c = cosb[(size_t)s * 64 + i], sn = sinb[(size_t)s * 64 + i];
    float x1 = q[2 * p], x2 = q[2 * p + 1];
    q[2 * p] = x1 * c - x2 * sn;  q[2 * p + 1] = x2 * c + x1 * sn;
    x1 = k[2 * p]; x2 = k[2 * p + 1];
    k[2 * p] = x1 * c - x2 * sn;  k[2 * p + 1] = x2 * c + x1 * sn;
  }

  const size_t qo = ((size_t)h * SEQ + s) * DHD + d0;
#pragma unroll
  for (int t = 0; t < 4; ++t) { Qm[qo + t] = (bf16)q[t]; Km[qo + t] = (bf16)k[t]; }
#pragma unroll
  for (int t = 0; t < 4; ++t)
    Vt[((size_t)h * DHD + d0 + t) * SEQ + s] = (bf16)v[t];
}

// ------------------------------------------------------------ flash attention
// One wave per (head, 16-row Q tile).  Keys streamed in tiles of 32.
__global__ __launch_bounds__(128) void flash_attn(
    const bf16* __restrict__ Qm, const bf16* __restrict__ Km,
    const bf16* __restrict__ Vt, bf16* __restrict__ O) {
  __shared__ bf16 Pl[4][16][32];  // per-wave P relayout scratch
  const int lane = threadIdx.x & 31, wv = threadIdx.x >> 5;
  const int l16 = lane & 15, hi = lane >> 4;
  const int task = blockIdx.x * 4 + wv;
  const int h = task / (SEQ / 16);
  const int q0 = (task % (SEQ / 16)) * 16;

  v16bf qa[4];
  {
    const bf16* qp = Qm + ((size_t)h * SEQ + q0 + l16) * DHD;
#pragma unroll
    for (int c = 0; c < 4; ++c)
      qa[c] = cat8(*(const v8bf*)(qp + c * 32 + hi * 16),
                   *(const v8bf*)(qp + c * 32 + hi * 16 + 8));
  }

  v8f acc[8];
#pragma unroll
  for (int f = 0; f < 8; ++f) acc[f] = vzero8();
  float m[8], l[8];
#pragma unroll
  for (int r = 0; r < 8; ++r) { m[r] = -1e30f; l[r] = 0.0f; }

  for (int kb = 0; kb < SEQ; kb += 32) {
    v8f s0 = vzero8(), s1 = vzero8();
#pragma unroll
    for (int c = 0; c < 4; ++c) {
      const bf16* kp0 = Km + ((size_t)h * SEQ + kb + l16) * DHD + c * 32 + hi * 16;
      v16bf b0 = cat8(*(const v8bf*)kp0, *(const v8bf*)(kp0 + 8));
      s0 = __builtin_amdgcn_wmma_f32_16x16x32_bf16(false, qa[c], false, b0,
                                                   (short)0, s0, false, false);
      const bf16* kp1 = kp0 + (size_t)16 * DHD;
      v16bf b1 = cat8(*(const v8bf*)kp1, *(const v8bf*)(kp1 + 8));
      s1 = __builtin_amdgcn_wmma_f32_16x16x32_bf16(false, qa[c], false, b1,
                                                   (short)0, s1, false, false);
    }
    float p0[8], p1[8], al[8];
#pragma unroll
    for (int r = 0; r < 8; ++r) {
      p0[r] = s0[r] * SCALE; p1[r] = s1[r] * SCALE;
      float t = fmaxf(p0[r], p1[r]);
#pragma unroll
      for (int msk = 1; msk < 16; msk <<= 1) t = fmaxf(t, __shfl_xor(t, msk, 32));
      const float mn = fmaxf(m[r], t);
      al[r] = __expf(m[r] - mn);
      p0[r] = __expf(p0[r] - mn);
      p1[r] = __expf(p1[r] - mn);
      float rs = p0[r] + p1[r];
#pragma unroll
      for (int msk = 1; msk < 16; msk <<= 1) rs += __shfl_xor(rs, msk, 32);
      l[r] = l[r] * al[r] + rs;
      m[r] = mn;
    }
#pragma unroll
    for (int f = 0; f < 8; ++f)
#pragma unroll
      for (int r = 0; r < 8; ++r) acc[f][r] *= al[r];

    // D-fragment -> A-fragment relayout through wave-private LDS
#pragma unroll
    for (int r = 0; r < 8; ++r) {
      Pl[wv][hi * 8 + r][l16]      = (bf16)p0[r];
      Pl[wv][hi * 8 + r][16 + l16] = (bf16)p1[r];
    }
    v8bf plo, phi;
#pragma unroll
    for (int i = 0; i < 8; ++i) {
      plo[i] = Pl[wv][l16][hi * 16 + i];
      phi[i] = Pl[wv][l16][hi * 16 + 8 + i];
    }
    const v16bf pa = cat8(plo, phi);

#pragma unroll
    for (int f = 0; f < 8; ++f) {
      const bf16* vp = Vt + (((size_t)h * DHD + f * 16 + l16) * SEQ + kb + hi * 16);
      v16bf vb = cat8(*(const v8bf*)vp, *(const v8bf*)(vp + 8));
      acc[f] = __builtin_amdgcn_wmma_f32_16x16x32_bf16(false, pa, false, vb,
                                                       (short)0, acc[f], false, false);
    }
  }

  float inv[8];
#pragma unroll
  for (int r = 0; r < 8; ++r) inv[r] = 1.0f / l[r];
#pragma unroll
  for (int f = 0; f < 8; ++f)
#pragma unroll
    for (int r = 0; r < 8; ++r)
      O[(size_t)(q0 + hi * 8 + r) * DD + h * DHD + f * 16 + l16] =
          (bf16)(acc[f][r] * inv[r]);
}

// ---------------------------------------------------------------------- host
extern "C" void kernel_launch(void* const* d_in, const int* in_sizes, int n_in,
                              void* d_out, int out_size, void* d_ws, size_t ws_size,
                              hipStream_t stream) {
  (void)in_sizes; (void)n_in; (void)out_size; (void)ws_size;
  const float* hs        = (const float*)d_in[0];
  const float* enc       = (const float*)d_in[1];
  const float* cosb      = (const float*)d_in[2];
  const float* sinb      = (const float*)d_in[3];
  const float* w_qkv     = (const float*)d_in[4];
  const float* w_add_qkv = (const float*)d_in[5];
  const float* b_add_qkv = (const float*)d_in[6];
  const float* nq        = (const float*)d_in[7];
  const float* nk        = (const float*)d_in[8];
  const float* naq       = (const float*)d_in[9];
  const float* nak       = (const float*)d_in[10];
  const float* w_out     = (const float*)d_in[11];
  const float* b_out     = (const float*)d_in[12];
  const float* w_add_out = (const float*)d_in[13];
  const float* b_add_out = (const float*)d_in[14];
  float* out = (float*)d_out;

  char* p = (char*)d_ws;
  auto take = [&](size_t bytes) -> char* {
    char* r = p; p += (bytes + 255) & ~(size_t)255; return r;
  };
  bf16*  wqkv_b    = (bf16*)take((size_t)E3 * DD * 2);
  bf16*  waddqkv_b = (bf16*)take((size_t)E3 * DD * 2);
  bf16*  wout_b    = (bf16*)take((size_t)DD * DD * 2);
  bf16*  waddout_b = (bf16*)take((size_t)DD * DD * 2);
  bf16*  hs_b      = (bf16*)take((size_t)IMG * DD * 2);
  bf16*  enc_b     = (bf16*)take((size_t)TXT * DD * 2);
  float* qkv_i     = (float*)take((size_t)IMG * E3 * 4);
  float* qkv_t     = (float*)take((size_t)TXT * E3 * 4);
  bf16*  Qm        = (bf16*)take((size_t)NH * SEQ * DHD * 2);
  bf16*  Km        = (bf16*)take((size_t)NH * SEQ * DHD * 2);
  bf16*  Vt        = (bf16*)take((size_t)NH * SEQ * DHD * 2);
  bf16*  attn      = (bf16*)take((size_t)SEQ * DD * 2);

  cvt_bf16<<<2048, 256, 0, stream>>>(w_qkv,     wqkv_b,    (size_t)E3 * DD);
  cvt_bf16<<<2048, 256, 0, stream>>>(w_add_qkv, waddqkv_b, (size_t)E3 * DD);
  cvt_bf16<<<1024, 256, 0, stream>>>(w_out,     wout_b,    (size_t)DD * DD);
  cvt_bf16<<<1024, 256, 0, stream>>>(w_add_out, waddout_b, (size_t)DD * DD);
  cvt_bf16<<<1024, 256, 0, stream>>>(hs,        hs_b,      (size_t)IMG * DD);
  cvt_bf16<<<256, 256, 0, stream>>>(enc,        enc_b,     (size_t)TXT * DD);

  gemm_bf16<<<dim3(E3 / 128, IMG / 128), 256, 0, stream>>>(
      hs_b, wqkv_b, nullptr, qkv_i, IMG, E3, DD);
  gemm_bf16<<<dim3(E3 / 128, TXT / 128), 256, 0, stream>>>(
      enc_b, waddqkv_b, b_add_qkv, qkv_t, TXT, E3, DD);

  pack_qkv<<<(SEQ * NH) / 8, 256, 0, stream>>>(
      qkv_i, qkv_t, cosb, sinb, nq, nk, naq, nak, Qm, Km, Vt);

  flash_attn<<<(NH * (SEQ / 16)) / 4, 128, 0, stream>>>(Qm, Km, Vt, attn);

  gemm_bf16<<<dim3(DD / 128, IMG / 128), 256, 0, stream>>>(
      attn + (size_t)TXT * DD, wout_b, b_out, out, IMG, DD, DD);
  gemm_bf16<<<dim3(DD / 128, TXT / 128), 256, 0, stream>>>(
      attn, waddout_b, b_add_out, out + (size_t)IMG * DD, TXT, DD, DD);
}
